// AttentionConv2d_3350074491538
// MI455X (gfx1250) — matrix operand order repaired and checked by reading, not compiled, not run on hardware
//
#include <hip/hip_runtime.h>
#include <hip/hip_bf16.h>

// AttentionConv2d for MI455X (gfx1250, wave32, WMMA f16->f32)
// B=8, CIN=256, H=W=32 (HW=1024), DK=DV=256, NH=8, DKH=32, COUT=512
//
// Strategy: one-time pre-pass converts/transposes all operands into f16
// buffers laid out to match WMMA fragment order, so every fragment load in
// the hot loops is a pair of 128-bit loads feeding v_wmma_f32_16x16x32_f16.

typedef __attribute__((ext_vector_type(16))) _Float16 v16h;
typedef __attribute__((ext_vector_type(8)))  _Float16 v8h;
typedef __attribute__((ext_vector_type(8)))  float    v8f;

static __device__ __forceinline__ v8f wmma_f16(v16h a, v16h b, v8f c) {
  return __builtin_amdgcn_wmma_f32_16x16x32_f16(false, a, false, b, (short)0, c, false, false);
}
static __device__ __forceinline__ v16h cat16(v8h lo, v8h hi) {
  return __builtin_shufflevector(lo, hi, 0,1,2,3,4,5,6,7,8,9,10,11,12,13,14,15);
}
// A-fragment: K = hh*16 + {0..15}  (one contiguous run)
static __device__ __forceinline__ v16h ldA(const _Float16* p) {
  return cat16(*(const v8h*)p, *(const v8h*)(p + 8));
}
// B-fragment: K = hh*16 + {0..15}  (one contiguous run)
static __device__ __forceinline__ v16h ldB(const _Float16* p) {
  return cat16(*(const v8h*)p, *(const v8h*)(p + 8));
}

// ---------------------------------------------------------------------------
// Pre-pass kernels (one-time, ~11 MB total; everything stays L2-resident)
// ---------------------------------------------------------------------------
__global__ __launch_bounds__(256) void cvt_f32_f16(const float* __restrict__ in,
                                                   _Float16* __restrict__ out, int n) {
  for (int i = blockIdx.x * 256 + threadIdx.x; i < n; i += gridDim.x * 256)
    out[i] = (_Float16)in[i];
}

// rel tables [63,32] -> padded [64,32] f16, row 63 zeroed (kills in-loop predicate)
__global__ __launch_bounds__(256) void cvt_rel(const float* __restrict__ rw,
                                               const float* __restrict__ rh,
                                               _Float16* __restrict__ rw16,
                                               _Float16* __restrict__ rh16) {
  int i = blockIdx.x * 256 + threadIdx.x;            // 0..2047
  bool ok = (i >> 5) < 63;
  rw16[i] = ok ? (_Float16)rw[i] : (_Float16)0.f;
  rh16[i] = ok ? (_Float16)rh[i] : (_Float16)0.f;
}

// x [b,256c,1024p] f32 -> xT [b,1024p,256c] f16 (B-fragment order for GEMM/conv)
__global__ __launch_bounds__(256) void transpose_x(const float* __restrict__ x,
                                                   _Float16* __restrict__ xT) {
  size_t i = (size_t)blockIdx.x * 256 + threadIdx.x;  // 2,097,152
  int c = i & 255, p = (i >> 8) & 1023, b = i >> 18;
  xT[i] = (_Float16)x[((size_t)b * 256 + c) * 1024 + p];
}

// w_out [o,c,3,3] f32 -> wT [tap9][o][c] f16 (A-fragment order per tap)
__global__ __launch_bounds__(256) void transpose_wconv(const float* __restrict__ w,
                                                       _Float16* __restrict__ wT) {
  size_t i = (size_t)blockIdx.x * 256 + threadIdx.x;  // 589,824
  int c = i & 255, o = (i >> 8) & 255, tap = i >> 16;
  wT[i] = (_Float16)w[((size_t)o * 256 + c) * 9 + tap];
}

// ---------------------------------------------------------------------------
// Kernel 1: QKV GEMM. out routed to q^T (scaled), k^T, v in attention layouts.
//   q^T,k^T: [b, h, p, 32d]   v: [b, h, 32d, 1024p]
// One wave = 16(o) x 16(p) tile, 8 K-steps of 32.
// ---------------------------------------------------------------------------
__global__ __launch_bounds__(256) void gemm_qkv(
    const _Float16* __restrict__ Aw,   // [768, 256]
    const _Float16* __restrict__ xT,   // [b, 1024, 256]
    const float* __restrict__ bias,    // [768]
    _Float16* __restrict__ qT, _Float16* __restrict__ kT, _Float16* __restrict__ vv)
{
  const int wid = threadIdx.x >> 5, lane = threadIdx.x & 31;
  const int tile = blockIdx.x * 8 + wid;              // 24576 waves
  const int b = tile / 3072, rem = tile - b * 3072;   // 48*64 tiles per batch
  const int ot = rem >> 6, pt = rem & 63;
  const int m = lane & 15, hh = lane >> 4;

  const _Float16* __restrict__ arow = Aw + (ot * 16 + m) * 256 + hh * 16;
  const _Float16* __restrict__ brow = xT + ((size_t)b * 1024 + pt * 16 + m) * 256 + hh * 16;
  v8f acc = {};
  #pragma unroll 1
  for (int c0 = 0; c0 < 256; c0 += 32)
    acc = wmma_f16(ldA(arow + c0), ldB(brow + c0), acc);

  const float scale = 0.17677669529663687f;           // 1/sqrt(32)
  const int p = pt * 16 + m;
  #pragma unroll
  for (int r = 0; r < 8; ++r) {
    const int o = ot * 16 + r + 8 * hh;
    const float val = acc[r] + bias[o];
    const int region = o >> 8, oc = o & 255, hd = oc >> 5, d = oc & 31;
    const size_t bh = (size_t)b * 8 + hd;
    if      (region == 0) qT[(bh * 1024 + p) * 32 + d] = (_Float16)(val * scale);
    else if (region == 1) kT[(bh * 1024 + p) * 32 + d] = (_Float16)val;
    else                  vv[(bh * 32 + d) * 1024 + p] = (_Float16)val;
  }
}

// ---------------------------------------------------------------------------
// Kernel 2: flash attention + relative logits. One wave per (b,h,16-query tile).
// logits[(x,y),(x2,y2)] = q.k + q.relw[y2-y+31] + q.relh[x2-x+31]
// ---------------------------------------------------------------------------
__global__ __launch_bounds__(128) void attn_flash(
    const _Float16* __restrict__ qT,     // [b,h,1024,32] (pre-scaled)
    const _Float16* __restrict__ kT,     // [b,h,1024,32]
    const _Float16* __restrict__ vv,     // [b,h,32,1024]
    const _Float16* __restrict__ relw,   // [64,32] padded
    const _Float16* __restrict__ relh,   // [64,32] padded
    _Float16* __restrict__ attnT)        // [b,1024,256]
{
  __shared__ float    s_rw[4][16 * 64];
  __shared__ float    s_rh[4][16 * 64];
  __shared__ _Float16 s_p [4][16 * 32];

  const int wid = threadIdx.x >> 5, lane = threadIdx.x & 31;
  const int g = blockIdx.x * 4 + wid;                 // 0..4095
  const int b = g >> 9, rem = g & 511, h = rem >> 6, qt = rem & 63;
  const int m = lane & 15, hh = lane >> 4;
  const size_t bh = (size_t)b * 8 + h;

  const _Float16* __restrict__ kbh = kT + bh * 1024 * 32;
  const _Float16* __restrict__ vbh = vv + bh * 32 * 1024;

  // Q A-fragment (reused for S / RW / RH): M=query, K=d
  const v16h qa = ldA(qT + (bh * 1024 + qt * 16 + m) * 32 + hh * 16);

  // Relative logits RW/RH = Q(16x32) x rel^T(32x64) -> LDS
  #pragma unroll 1
  for (int mt = 0; mt < 4; ++mt) {
    v8f z = {};
    v8f cw = wmma_f16(qa, ldB(relw + (mt * 16 + m) * 32 + hh * 16), z);
    v8f ch = wmma_f16(qa, ldB(relh + (mt * 16 + m) * 32 + hh * 16), z);
    #pragma unroll
    for (int r = 0; r < 8; ++r) {
      s_rw[wid][(r + 8 * hh) * 64 + mt * 16 + m] = cw[r];
      s_rh[wid][(r + 8 * hh) * 64 + mt * 16 + m] = ch[r];
    }
  }
  __syncthreads();

  v8f o0 = {}, o1 = {};
  float mrun[8], lrun[8];
  #pragma unroll
  for (int r = 0; r < 8; ++r) { mrun[r] = -1e30f; lrun[r] = 0.f; }

  #pragma unroll 1
  for (int ch = 0; ch < 32; ++ch) {
    const int k0 = ch * 32;
    v8f z = {};
    v8f S0 = wmma_f16(qa, ldB(kbh + (size_t)(k0 + m)      * 32 + hh * 16), z);
    v8f S1 = wmma_f16(qa, ldB(kbh + (size_t)(k0 + 16 + m) * 32 + hh * 16), z);

    const int gk0 = k0 + m, gk1 = k0 + 16 + m;
    const int x20 = gk0 >> 5, y20 = gk0 & 31, x21 = gk1 >> 5, y21 = gk1 & 31;

    #pragma unroll
    for (int r = 0; r < 8; ++r) {
      const int qrow = r + 8 * hh;
      const int gq = qt * 16 + qrow;
      const int qx = gq >> 5, qy = gq & 31;
      S0[r] += s_rw[wid][qrow * 64 + (y20 - qy + 31)] + s_rh[wid][qrow * 64 + (x20 - qx + 31)];
      S1[r] += s_rw[wid][qrow * 64 + (y21 - qy + 31)] + s_rh[wid][qrow * 64 + (x21 - qx + 31)];

      float mx = fmaxf(S0[r], S1[r]);
      mx = fmaxf(mx, __shfl_xor(mx, 1));
      mx = fmaxf(mx, __shfl_xor(mx, 2));
      mx = fmaxf(mx, __shfl_xor(mx, 4));
      mx = fmaxf(mx, __shfl_xor(mx, 8));
      const float mnew = fmaxf(mrun[r], mx);
      const float corr = __expf(mrun[r] - mnew);
      const float p0 = __expf(S0[r] - mnew);
      const float p1 = __expf(S1[r] - mnew);
      float ps = p0 + p1;
      ps += __shfl_xor(ps, 1);
      ps += __shfl_xor(ps, 2);
      ps += __shfl_xor(ps, 4);
      ps += __shfl_xor(ps, 8);
      lrun[r] = lrun[r] * corr + ps;
      mrun[r] = mnew;
      o0[r] *= corr;
      o1[r] *= corr;
      s_p[wid][qrow * 32 + m]      = (_Float16)p0;
      s_p[wid][qrow * 32 + 16 + m] = (_Float16)p1;
    }
    __syncthreads();

    // O += P(16x32) x V^T(32x16) per d-half; P read back in A-layout (ds_b128)
    const v16h pa = ldA((const _Float16*)&s_p[wid][m * 32 + hh * 16]);
    o0 = wmma_f16(pa, ldB(vbh + (size_t)m        * 1024 + k0 + hh * 16), o0);
    o1 = wmma_f16(pa, ldB(vbh + (size_t)(16 + m) * 1024 + k0 + hh * 16), o1);
    __syncthreads();
  }

  // normalize + store f16 in [p, c] order (B-fragment order for final proj)
  #pragma unroll
  for (int r = 0; r < 8; ++r) {
    const int gq = qt * 16 + r + 8 * hh;
    const float inv = 1.0f / lrun[r];
    _Float16* dst = attnT + ((size_t)b * 1024 + gq) * 256 + h * 32;
    dst[m]      = (_Float16)(o0[r] * inv);
    dst[16 + m] = (_Float16)(o1[r] * inv);
  }
}

// ---------------------------------------------------------------------------
// Kernel 3: final 1x1 projection into d_out channels 256..511 (f32 out)
// ---------------------------------------------------------------------------
__global__ __launch_bounds__(256) void gemm_proj(
    const _Float16* __restrict__ Aw,   // [256, 256]
    const _Float16* __restrict__ X,    // [b, 1024, 256]
    const float* __restrict__ bias,    // [256]
    float* __restrict__ out)           // [b, 512, 1024]
{
  const int wid = threadIdx.x >> 5, lane = threadIdx.x & 31;
  const int tile = blockIdx.x * 8 + wid;              // 8192 waves
  const int b = tile >> 10, rem = tile & 1023;        // 16*64 tiles per batch
  const int ot = rem >> 6, pt = rem & 63;
  const int m = lane & 15, hh = lane >> 4;

  const _Float16* __restrict__ arow = Aw + (ot * 16 + m) * 256 + hh * 16;
  const _Float16* __restrict__ brow = X + ((size_t)b * 1024 + pt * 16 + m) * 256 + hh * 16;
  v8f acc = {};
  #pragma unroll 1
  for (int c0 = 0; c0 < 256; c0 += 32)
    acc = wmma_f16(ldA(arow + c0), ldB(brow + c0), acc);

  #pragma unroll
  for (int r = 0; r < 8; ++r) {
    const int o = ot * 16 + r + 8 * hh;
    out[((size_t)b * 512 + 256 + o) * 1024 + pt * 16 + m] = acc[r] + bias[o];
  }
}

// ---------------------------------------------------------------------------
// Kernel 4: 3x3 conv (pad 1) as implicit GEMM. 9 taps x 8 K-chunks = 72 WMMAs.
// ---------------------------------------------------------------------------
__global__ __launch_bounds__(256) void conv3x3(
    const _Float16* __restrict__ xT,   // [b, 1024(y,x), 256c]
    const _Float16* __restrict__ wT,   // [tap9, 256o, 256c]
    const float* __restrict__ bias,    // [256]
    float* __restrict__ out)           // [b, 512, 1024] (channels 0..255)
{
  const int wid = threadIdx.x >> 5, lane = threadIdx.x & 31;
  const int tile = blockIdx.x * 8 + wid;              // 8192 waves
  const int b = tile >> 10, rem = tile & 1023;
  const int ot = rem >> 6, pt = rem & 63;
  const int y = pt >> 1, x0 = (pt & 1) * 16;
  const int m = lane & 15, hh = lane >> 4;

  const v16h vzero = {};
  v8f acc = {};

  #pragma unroll 1
  for (int kh = 0; kh < 3; ++kh) {
    const int iy = y + kh - 1;
    if (iy < 0 || iy >= 32) continue;                 // wave-uniform branch
    #pragma unroll 1
    for (int kw = 0; kw < 3; ++kw) {
      const int ix = x0 + m + kw - 1;
      const bool okx = (ix >= 0) && (ix < 32);        // per-lane data predicate
      const int ixc = ix < 0 ? 0 : (ix > 31 ? 31 : ix);
      const _Float16* __restrict__ arow =
          wT + ((size_t)(kh * 3 + kw) * 256 + ot * 16 + m) * 256 + hh * 16;
      const _Float16* __restrict__ brow =
          xT + ((size_t)b * 1024 + iy * 32 + ixc) * 256 + hh * 16;
      #pragma unroll 1
      for (int c0 = 0; c0 < 256; c0 += 32) {
        v16h bb = ldB(brow + c0);
        bb = okx ? bb : vzero;                        // zero-pad via select
        acc = wmma_f16(ldA(arow + c0), bb, acc);
      }
    }
  }

  #pragma unroll
  for (int r = 0; r < 8; ++r) {
    const int o = ot * 16 + r + 8 * hh;
    out[((size_t)b * 512 + o) * 1024 + y * 32 + x0 + m] = acc[r] + bias[o];
  }
}

// ---------------------------------------------------------------------------
extern "C" void kernel_launch(void* const* d_in, const int* in_sizes, int n_in,
                              void* d_out, int out_size, void* d_ws, size_t ws_size,
                              hipStream_t stream) {
  const float* x      = (const float*)d_in[0];
  const float* w_qkv  = (const float*)d_in[1];
  const float* b_qkv  = (const float*)d_in[2];
  const float* w_attn = (const float*)d_in[3];
  const float* b_attn = (const float*)d_in[4];
  const float* w_out  = (const float*)d_in[5];
  const float* b_out  = (const float*)d_in[6];
  const float* krw    = (const float*)d_in[7];
  const float* krh    = (const float*)d_in[8];
  float* out = (float*)d_out;

  // f16 workspace layout (~22.7 MB, L2-resident)
  _Float16* hp    = (_Float16*)d_ws;
  _Float16* xT    = hp;                          // 8*1024*256   = 2,097,152
  _Float16* wq16  = xT   + (size_t)2097152;      // 768*256      =   196,608
  _Float16* wa16  = wq16 + (size_t)196608;       // 256*256      =    65,536
  _Float16* wcT   = wa16 + (size_t)65536;        // 9*256*256    =   589,824
  _Float16* rw16  = wcT  + (size_t)589824;       // 64*32        =     2,048
  _Float16* rh16  = rw16 + (size_t)2048;         // 64*32        =     2,048
  _Float16* qT    = rh16 + (size_t)2048;         // 8*8*1024*32  = 2,097,152
  _Float16* kT    = qT   + (size_t)2097152;
  _Float16* vv    = kT   + (size_t)2097152;
  _Float16* attnT = vv   + (size_t)2097152;      // 8*1024*256   = 2,097,152

  // --- one-time operand staging (convert + transpose into fragment order) ---
  transpose_x   <<<8192, 256, 0, stream>>>(x, xT);
  cvt_f32_f16   <<<768,  256, 0, stream>>>(w_qkv,  wq16, 768 * 256);
  cvt_f32_f16   <<<256,  256, 0, stream>>>(w_attn, wa16, 256 * 256);
  transpose_wconv<<<2304, 256, 0, stream>>>(w_out, wcT);
  cvt_rel       <<<8,    256, 0, stream>>>(krw, krh, rw16, rh16);

  // --- compute ---
  conv3x3  <<<1024, 256, 0, stream>>>(xT, wcT, b_out, out);
  gemm_qkv <<<3072, 256, 0, stream>>>(wq16, xT, b_qkv, qT, kT, vv);
  attn_flash<<<1024, 128, 0, stream>>>(qT, kT, vv, rw16, rh16, attnT);
  gemm_proj<<<1024, 256, 0, stream>>>(wa16, attnT, b_attn, out);
}
